// decoder_29326036697123
// MI455X (gfx1250) — compile-verified, matrix operand order checked by reading, not run on hardware
//
#include <hip/hip_runtime.h>
#include <hip/hip_bf16.h>
#include <math.h>

// ---------------------------------------------------------------------------
// GRU teacher-forced decode for MI455X (gfx1250, wave32, WMMA).
//   Phase 1 (tensor cores): gi[T,3H] = [sos; y[:-1]] @ W_ih^T + b_ih
//     - LDS double-buffered tiles fed by async global->LDS (ASYNCcnt) copies
//   Phase 2 (persistent):   8192-step scan, W_hh slice LDS-resident per block
// ---------------------------------------------------------------------------

typedef __attribute__((ext_vector_type(16))) _Float16 v16h;
typedef __attribute__((ext_vector_type(8)))  _Float16 v8h;
typedef __attribute__((ext_vector_type(8)))  float    v8f;

#define T_STEPS 8192
#define HDIM    1024
#define H3      3072

// -------------------- workspace layout (bytes, 256-aligned) ----------------
#define OFF_INPH  ((size_t)0)                         // _Float16[T*H]   16 MB
#define OFF_WIHH  (OFF_INPH + (size_t)T_STEPS*HDIM*2) // _Float16[3H*H]   6 MB
#define OFF_GI    (OFF_WIHH + (size_t)H3*HDIM*2)      // float[T*3H]     96 MB
#define OFF_HBUF  (OFF_GI   + (size_t)T_STEPS*H3*4)   // float[2*H]       8 KB
#define OFF_BAR   (OFF_HBUF + (size_t)2*HDIM*4)       // int[2]

// ----------------- CDNA5 async global->LDS helpers (ISA ch.10/15.18) -------
__device__ __forceinline__ unsigned lds_off(const void* p) {
    // addrspace(3) offset == low 32 bits of the generic pointer
    return (unsigned)(uintptr_t)p;
}
__device__ __forceinline__ void async_load_b128(unsigned dst_lds, const void* src) {
    asm volatile("global_load_async_to_lds_b128 %0, %1, off"
                 :: "v"(dst_lds), "v"(src) : "memory");
}
__device__ __forceinline__ void wait_asynccnt0() {
    asm volatile("s_wait_asynccnt 0" ::: "memory");
}

// --------------------------- init: h0 = x, barrier = 0 ---------------------
__global__ void gru_init_kernel(const float* __restrict__ x,
                                float* __restrict__ hbuf,
                                int* __restrict__ bar) {
    int i = blockIdx.x * blockDim.x + threadIdx.x;
    if (i < HDIM) {
        hbuf[i]        = x[i];
        hbuf[HDIM + i] = 0.0f;
    }
    if (i == 0) { bar[0] = 0; bar[1] = 0; }
}

// ------------------- fp32 -> f16 conversion (inp shift + W_ih) -------------
__global__ void gru_cvt_kernel(const float* __restrict__ sos,
                               const float* __restrict__ y,
                               const float* __restrict__ Wih,
                               _Float16* __restrict__ inpH,
                               _Float16* __restrict__ WihH) {
    size_t stride = (size_t)gridDim.x * blockDim.x;
    size_t i0 = (size_t)blockIdx.x * blockDim.x + threadIdx.x;

    const size_t n_inp = (size_t)T_STEPS * HDIM;
    for (size_t idx = i0; idx < n_inp; idx += stride) {
        size_t t = idx >> 10;          // / HDIM
        size_t c = idx & (HDIM - 1);   // % HDIM
        float v = (t == 0) ? sos[c] : y[(t - 1) * HDIM + c];
        inpH[idx] = (_Float16)v;
    }
    const size_t n_w = (size_t)H3 * HDIM;
    for (size_t idx = i0; idx < n_w; idx += stride) {
        WihH[idx] = (_Float16)Wih[idx];
    }
}

// -------------------- WMMA GEMM: gi = inp @ W_ih^T + b_ih ------------------
// Block: 256 threads = 8 waves (4 M x 2 N). Block tile 128M x 64N, K-slab 32.
// LDS double buffer: A 128x32 f16 (8 KB) + B 64x32 f16 (4 KB), x2 = 24 KB.
// Staging: async global->LDS b128; 3 async instructions per wave per slab,
// overlapped with WMMA on the other buffer; drained by s_wait_asynccnt 0.
__global__ __launch_bounds__(256)
void gru_gemm_kernel(const _Float16* __restrict__ inpH,
                     const _Float16* __restrict__ WihH,
                     const float* __restrict__ bih,
                     float* __restrict__ gi) {
    __shared__ _Float16 sA[2][128 * 32];
    __shared__ _Float16 sB[2][64 * 32];

    const int tid  = threadIdx.x;
    const int lane = tid & 31;
    const int wave = tid >> 5;
    const int mw   = wave & 3;   // 0..3
    const int nw   = wave >> 2;  // 0..1

    const int blockM = blockIdx.x * 128;
    const int blockN = blockIdx.y * 64;

    // --- staging map: 16B transfers; A: e=tid, tid+256; B: e=tid ---
    const int rA0 = tid >> 2,            kA0 = (tid & 3) * 8;
    const int rA1 = (tid + 256) >> 2,    kA1 = kA0;           // (e&3) identical
    const int rB  = tid >> 2,            kB  = (tid & 3) * 8; // rB in 0..63
    const _Float16* gA0 = inpH + (size_t)(blockM + rA0) * HDIM + kA0;
    const _Float16* gA1 = inpH + (size_t)(blockM + rA1) * HDIM + kA1;
    const _Float16* gB  = WihH + (size_t)(blockN + rB)  * HDIM + kB;
    const unsigned dA0[2] = { lds_off(&sA[0][rA0 * 32 + kA0]),
                              lds_off(&sA[1][rA0 * 32 + kA0]) };
    const unsigned dA1[2] = { lds_off(&sA[0][rA1 * 32 + kA1]),
                              lds_off(&sA[1][rA1 * 32 + kA1]) };
    const unsigned dB[2]  = { lds_off(&sB[0][rB * 32 + kB]),
                              lds_off(&sB[1][rB * 32 + kB]) };

    // --- fragment map (16-bit A/B striping: lane&15 = row/col, lane>>4 = K half)
    const int lrow = lane & 15;
    const int lk16 = (lane >> 4) * 16;
    const int aRow0 = mw * 32 + lrow;        // M fragment rows in block tile
    const int aRow1 = mw * 32 + 16 + lrow;
    const int bRow0 = nw * 32 + lrow;        // N fragment cols in block tile
    const int bRow1 = nw * 32 + 16 + lrow;

    v8f acc[2][2] = {{{0.f,0.f,0.f,0.f,0.f,0.f,0.f,0.f},
                      {0.f,0.f,0.f,0.f,0.f,0.f,0.f,0.f}},
                     {{0.f,0.f,0.f,0.f,0.f,0.f,0.f,0.f},
                      {0.f,0.f,0.f,0.f,0.f,0.f,0.f,0.f}}};

    // prologue: stage slab 0
    async_load_b128(dA0[0], gA0);
    async_load_b128(dA1[0], gA1);
    async_load_b128(dB[0],  gB);
    wait_asynccnt0();
    __syncthreads();

    #pragma unroll 2
    for (int ks = 0; ks < 32; ++ks) {
        const int cur = ks & 1;
        if (ks + 1 < 32) {   // stage next slab while computing this one
            const int nk = (ks + 1) * 32;
            async_load_b128(dA0[cur ^ 1], gA0 + nk);
            async_load_b128(dA1[cur ^ 1], gA1 + nk);
            async_load_b128(dB[cur ^ 1],  gB  + nk);
        }

        const v16h a0 = *(const v16h*)&sA[cur][aRow0 * 32 + lk16];
        const v16h a1 = *(const v16h*)&sA[cur][aRow1 * 32 + lk16];
        const v16h b0 = *(const v16h*)&sB[cur][bRow0 * 32 + lk16];
        const v16h b1 = *(const v16h*)&sB[cur][bRow1 * 32 + lk16];

        acc[0][0] = __builtin_amdgcn_wmma_f32_16x16x32_f16(
            false, a0, false, b0, (short)0, acc[0][0], false, false);
        acc[0][1] = __builtin_amdgcn_wmma_f32_16x16x32_f16(
            false, a0, false, b1, (short)0, acc[0][1], false, false);
        acc[1][0] = __builtin_amdgcn_wmma_f32_16x16x32_f16(
            false, a1, false, b0, (short)0, acc[1][0], false, false);
        acc[1][1] = __builtin_amdgcn_wmma_f32_16x16x32_f16(
            false, a1, false, b1, (short)0, acc[1][1], false, false);

        wait_asynccnt0();     // next slab landed
        __syncthreads();      // everyone done reading cur before it's reused
    }

    // C/D layout: lane holds N = (lane&15); VGPR r holds M = r + (lane>>4)*8.
    const int mhalf = (lane >> 4) * 8;
    #pragma unroll
    for (int ni = 0; ni < 2; ++ni) {
        const int ncol = blockN + nw * 32 + ni * 16 + (lane & 15);
        const float bias = bih[ncol];
        #pragma unroll
        for (int mi = 0; mi < 2; ++mi) {
            #pragma unroll
            for (int r = 0; r < 8; ++r) {
                const int mrow = blockM + mw * 32 + mi * 16 + mhalf + r;
                gi[(size_t)mrow * H3 + ncol] = acc[mi][ni][r] + bias;
            }
        }
    }
}

// ------------- persistent sequential scan: h_t = GRUCell(gi_t, h) ----------
// 64 blocks x 256 threads; block owns 16 hidden units j -> 48 W_hh rows,
// converted once to f16 and kept LDS-resident (96 KB of the 320 KB WGP LDS).
// Per step: h (4 KB) -> LDS, 6 dots/wave from LDS, shuffle-reduce, gates,
// ping-pong h through global, monotone-generation grid barrier.
#define SCAN_NB   64
#define J_PER_BLK 16            // HDIM / SCAN_NB
#define W_ROWS    (3 * J_PER_BLK)
__global__ __launch_bounds__(256)
void gru_scan_kernel(const float* __restrict__ Whh,
                     const float* __restrict__ bhh,
                     const float* __restrict__ gi,
                     float* __restrict__ out,
                     float* __restrict__ hbuf,
                     int* __restrict__ bar) {
    __shared__ _Float16 wlds[W_ROWS * HDIM];   // [lj*3+g][k], 96 KB
    __shared__ float    hs[HDIM];              // 4 KB

    const int tid  = threadIdx.x;
    const int lane = tid & 31;
    const int wave = tid >> 5;
    const int jblk = blockIdx.x * J_PER_BLK;

    // one-time: convert this block's W_hh slice to f16 in LDS
    for (int idx = tid; idx < W_ROWS * HDIM; idx += 256) {
        const int lr = idx >> 10;          // local row 0..47
        const int k  = idx & (HDIM - 1);
        const int lj = lr / 3;
        const int g  = lr - 3 * lj;
        wlds[idx] = (_Float16)Whh[(size_t)(g * HDIM + jblk + lj) * HDIM + k];
    }

    int* cnt = bar;
    int* gen = bar + 1;
    const int ljbase = wave * 2;               // 2 hidden units per wave

    for (int t = 0; t < T_STEPS; ++t) {
        const float* __restrict__ hcur  = hbuf + (t & 1) * HDIM;
        float* __restrict__       hnext = hbuf + ((t + 1) & 1) * HDIM;
        const float* __restrict__ git   = gi + (size_t)t * H3;

        // pull h into LDS (256 threads x float4 = 4 KB)
        ((float4*)hs)[tid] = ((const float4*)hcur)[tid];
        __syncthreads();

        float acc[2][3] = {{0.f,0.f,0.f},{0.f,0.f,0.f}};
        #pragma unroll
        for (int p = 0; p < 4; ++p) {
            const int k = p * 256 + lane * 8;
            const float4 ha = *(const float4*)&hs[k];
            const float4 hb = *(const float4*)&hs[k + 4];
            const float hv[8] = { ha.x, ha.y, ha.z, ha.w, hb.x, hb.y, hb.z, hb.w };
            #pragma unroll
            for (int lj = 0; lj < 2; ++lj) {
                #pragma unroll
                for (int g = 0; g < 3; ++g) {
                    const v8h w = *(const v8h*)&wlds[((ljbase + lj) * 3 + g) * HDIM + k];
                    #pragma unroll
                    for (int i = 0; i < 8; ++i)
                        acc[lj][g] = fmaf((float)w[i], hv[i], acc[lj][g]);
                }
            }
        }
        #pragma unroll
        for (int off = 16; off > 0; off >>= 1) {
            #pragma unroll
            for (int lj = 0; lj < 2; ++lj) {
                acc[lj][0] += __shfl_xor(acc[lj][0], off, 32);
                acc[lj][1] += __shfl_xor(acc[lj][1], off, 32);
                acc[lj][2] += __shfl_xor(acc[lj][2], off, 32);
            }
        }
        if (lane == 0) {
            #pragma unroll
            for (int lj = 0; lj < 2; ++lj) {
                const int j = jblk + ljbase + lj;
                const float r  = 1.f / (1.f + __expf(-(git[j]            + acc[lj][0] + bhh[j])));
                const float z  = 1.f / (1.f + __expf(-(git[HDIM + j]     + acc[lj][1] + bhh[HDIM + j])));
                const float nn = tanhf(git[2 * HDIM + j] + r * (acc[lj][2] + bhh[2 * HDIM + j]));
                const float hnew = (1.f - z) * nn + z * hs[j];
                hnext[j] = hnew;
                out[(size_t)t * HDIM + j] = hnew;
            }
        }

        // ---- grid-wide barrier (two-phase, monotone generation == t+1) ----
        __threadfence();
        __syncthreads();
        if (tid == 0) {
            if (atomicAdd(cnt, 1) == SCAN_NB - 1) {
                atomicExch(cnt, 0);
                __threadfence();
                atomicAdd(gen, 1);
            } else {
                while (atomicAdd(gen, 0) < t + 1) {
                    __builtin_amdgcn_s_sleep(1);
                }
            }
        }
        __syncthreads();
        __threadfence();
    }
}

// ---------------------------------------------------------------------------
extern "C" void kernel_launch(void* const* d_in, const int* in_sizes, int n_in,
                              void* d_out, int out_size, void* d_ws, size_t ws_size,
                              hipStream_t stream) {
    const float* x   = (const float*)d_in[0];   // [H]
    const float* sos = (const float*)d_in[1];   // [H]
    const float* y   = (const float*)d_in[2];   // [T,H]
    const float* Wih = (const float*)d_in[3];   // [3H,H]
    const float* Whh = (const float*)d_in[4];   // [3H,H]
    const float* bih = (const float*)d_in[5];   // [3H]
    const float* bhh = (const float*)d_in[6];   // [3H]
    float* out = (float*)d_out;                 // [T,H]

    char* ws = (char*)d_ws;
    _Float16* inpH = (_Float16*)(ws + OFF_INPH);
    _Float16* WihH = (_Float16*)(ws + OFF_WIHH);
    float*    gi   = (float*)   (ws + OFF_GI);
    float*    hbuf = (float*)   (ws + OFF_HBUF);
    int*      bar  = (int*)     (ws + OFF_BAR);

    // 1) init barrier + h0
    gru_init_kernel<<<4, 256, 0, stream>>>(x, hbuf, bar);

    // 2) fp32 -> f16 staging
    gru_cvt_kernel<<<4096, 256, 0, stream>>>(sos, y, Wih, inpH, WihH);

    // 3) WMMA GEMM: gi = inp @ W_ih^T + b_ih
    dim3 ggrid(T_STEPS / 128, H3 / 64);
    gru_gemm_kernel<<<ggrid, 256, 0, stream>>>(inpH, WihH, bih, gi);

    // 4) persistent sequential GRU scan (LDS-resident W_hh slices)
    gru_scan_kernel<<<SCAN_NB, 256, 0, stream>>>(Whh, bhh, gi, out, hbuf, bar);
}
